// GRUModel_4466765988087
// MI455X (gfx1250) — matrix operand order repaired and checked by reading, not compiled, not run on hardware
//
#include <hip/hip_runtime.h>

typedef __attribute__((ext_vector_type(16))) _Float16 v16h;
typedef __attribute__((ext_vector_type(8)))  _Float16 v8h;
typedef __attribute__((ext_vector_type(8)))  float    v8f;
typedef __attribute__((ext_vector_type(4)))  float    v4f;

#define HID 32
#define SEQT 512
#define OO 12

__device__ __forceinline__ float fast_tanh(float x) {
#if __has_builtin(__builtin_amdgcn_tanhf)
    return __builtin_amdgcn_tanhf(x);      // gfx1250 v_tanh_f32
#else
    float e = __expf(2.0f * x);
    return (e - 1.0f) / (e + 1.0f);
#endif
}
__device__ __forceinline__ float fast_sigmoid(float x) {
    return 0.5f + 0.5f * fast_tanh(0.5f * x);
}

// One wave32 owns 16 batch rows. Per step: gh[96x16] = w_hh[96x32] @ h[32x16]
// as 6x V_WMMA_F32_16X16X32_F16 (A = permuted w_hh gate tile, B = h).
//
// Gate-row permutation trick: for gate type gt (r,z,n) and tile parity p,
// tile tt=2*gt+p has row M -> h-index j = p*8 + (M&7) + (M>=8 ? 16 : 0).
// D layout gives lane (half hf) VGPR m -> row M = m + hf*8, i.e.
//   p=0: j = m + hf*16     (state hlo[m])
//   p=1: j = m + 8 + hf*16 (state hhi[m])
// so a lane in half hf owns exactly j in [hf*16, hf*16+16) -- the exact
// K-range that lane supplies in the f16 32x16 B fragment (lanes 0-15:
// K=0..15, lanes 16-31: K=16..31). WMMA output feeds the next step's B
// fragment with NO cross-lane exchange: just 8 v_cvt_pk_f16_f32.
__global__ __launch_bounds__(256, 1)
void gru_wmma_kernel(const float* __restrict__ x,
                     const float* __restrict__ w_ih,
                     const float* __restrict__ w_hh,
                     const float* __restrict__ b_ih,
                     const float* __restrict__ b_hh,
                     const float* __restrict__ fc_w,
                     const float* __restrict__ fc_b,
                     float* __restrict__ out)
{
    const int lane = threadIdx.x & 31;
    const int wave = (int)((blockIdx.x * blockDim.x + threadIdx.x) >> 5);
    const int bn   = lane & 15;     // B/C column = batch within tile; also A row
    const int hf   = lane >> 4;     // lane half
    const int b    = wave * 16 + bn;

    // ---- A fragments: permuted w_hh gate tiles (resident in VGPRs) ----
    // A-fragment K packing (16-bit A 16x32): idx0..15 -> K = hf*16+idx
    // (contiguous, matching the B fragment).
    v16h A[6];
    #pragma unroll
    for (int gt = 0; gt < 3; ++gt) {
        #pragma unroll
        for (int p = 0; p < 2; ++p) {
            const int jrow = p * 8 + (bn & 7) + ((bn >> 3) * 16); // permuted row
            const float* wr = w_hh + (gt * 32 + jrow) * HID;
            v16h a;
            #pragma unroll
            for (int i = 0; i < 8; ++i) {
                a[i]     = (_Float16)wr[hf * 16 + i];
                a[8 + i] = (_Float16)wr[hf * 16 + 8 + i];
            }
            A[gt * 2 + p] = a;
        }
    }

    // ---- per-lane gate constants, permuted: g = gt*32 + p*8 + m + hf*16 ----
    v8h  wihp[6];     // w_ih (I==1), packed f16
    v8f  cinit[6];    // WMMA C init: r/z get b_ih+b_hh; n gets b_hh only
    v8h  bihnp[2];    // b_ih for n gates, packed f16
    #pragma unroll
    for (int gt = 0; gt < 3; ++gt) {
        #pragma unroll
        for (int p = 0; p < 2; ++p) {
            const int tt = gt * 2 + p;
            #pragma unroll
            for (int m = 0; m < 8; ++m) {
                int g = gt * 32 + p * 8 + m + hf * 16;
                wihp[tt][m] = (_Float16)w_ih[g];
                if (gt < 2) {
                    cinit[tt][m] = b_ih[g] + b_hh[g];
                } else {
                    cinit[tt][m] = b_hh[g];
                    bihnp[p][m] = (_Float16)b_ih[g];
                }
            }
        }
    }

    // hidden state, lane-local: hlo[m] = h[m + hf*16], hhi[m] = h[m+8+hf*16]
    float hlo[8], hhi[8];
    #pragma unroll
    for (int i = 0; i < 8; ++i) { hlo[i] = 0.0f; hhi[i] = 0.0f; }

    const float* xp = x + (size_t)b * SEQT;

    #pragma unroll 1
    for (int t0 = 0; t0 < SEQT; t0 += 4) {
        v4f xa = *reinterpret_cast<const v4f*>(xp + t0);
        float xbuf[4] = {xa[0], xa[1], xa[2], xa[3]};

        #pragma unroll
        for (int ts = 0; ts < 4; ++ts) {
            float xv = xbuf[ts];

            // B fragment (f16 32x16): lane col N=bn, K = hf*16 + idx.
            // Lane-local by construction -- no shuffles.
            v16h Bf;
            #pragma unroll
            for (int i = 0; i < 8; ++i) {
                Bf[i]     = (_Float16)hlo[i];
                Bf[8 + i] = (_Float16)hhi[i];
            }

            v8f D0 = __builtin_amdgcn_wmma_f32_16x16x32_f16(false, A[0], false, Bf, (short)0, cinit[0], false, false);
            v8f D1 = __builtin_amdgcn_wmma_f32_16x16x32_f16(false, A[1], false, Bf, (short)0, cinit[1], false, false);
            v8f D2 = __builtin_amdgcn_wmma_f32_16x16x32_f16(false, A[2], false, Bf, (short)0, cinit[2], false, false);
            v8f D3 = __builtin_amdgcn_wmma_f32_16x16x32_f16(false, A[3], false, Bf, (short)0, cinit[3], false, false);
            v8f D4 = __builtin_amdgcn_wmma_f32_16x16x32_f16(false, A[4], false, Bf, (short)0, cinit[4], false, false);
            v8f D5 = __builtin_amdgcn_wmma_f32_16x16x32_f16(false, A[5], false, Bf, (short)0, cinit[5], false, false);

            // GRU elementwise update (lane-local, fp32 state)
            #pragma unroll
            for (int i = 0; i < 8; ++i) {
                float rl = fast_sigmoid(fmaf(xv, (float)wihp[0][i], D0[i]));
                float rh = fast_sigmoid(fmaf(xv, (float)wihp[1][i], D1[i]));
                float zl = fast_sigmoid(fmaf(xv, (float)wihp[2][i], D2[i]));
                float zh = fast_sigmoid(fmaf(xv, (float)wihp[3][i], D3[i]));
                float nl = fast_tanh(fmaf(xv, (float)wihp[4][i], (float)bihnp[0][i]) + rl * D4[i]);
                float nh = fast_tanh(fmaf(xv, (float)wihp[5][i], (float)bihnp[1][i]) + rh * D5[i]);
                hlo[i] = fmaf(zl, hlo[i] - nl, nl);   // (1-z)*n + z*h
                hhi[i] = fmaf(zh, hhi[i] - nh, nh);
            }
        }
    }

    // ---- final FC: out[12 x 16] = fc_w @ h_last + fc_b, one WMMA ----
    // Build fc fragments only now, so they don't stay live across the loop.
    v16h Afc;
    #pragma unroll
    for (int i = 0; i < 8; ++i) {
        float lo = (bn < OO) ? fc_w[bn * HID + hf * 16 + i]     : 0.0f;
        float hi = (bn < OO) ? fc_w[bn * HID + hf * 16 + 8 + i] : 0.0f;
        Afc[i]     = (_Float16)lo;
        Afc[8 + i] = (_Float16)hi;
    }
    v8f fcc;
    #pragma unroll
    for (int m = 0; m < 8; ++m) {
        int o = hf * 8 + m;
        fcc[m] = (o < OO) ? fc_b[o] : 0.0f;
    }
    v16h Bf;
    #pragma unroll
    for (int i = 0; i < 8; ++i) {
        Bf[i]     = (_Float16)hlo[i];
        Bf[8 + i] = (_Float16)hhi[i];
    }
    v8f Do = __builtin_amdgcn_wmma_f32_16x16x32_f16(false, Afc, false, Bf, (short)0, fcc, false, false);

    #pragma unroll
    for (int m = 0; m < 8; ++m) {
        int o = hf * 8 + m;
        if (o < OO) out[(size_t)b * OO + o] = Do[m];
    }
}

extern "C" void kernel_launch(void* const* d_in, const int* in_sizes, int n_in,
                              void* d_out, int out_size, void* d_ws, size_t ws_size,
                              hipStream_t stream) {
    (void)in_sizes; (void)n_in; (void)out_size; (void)d_ws; (void)ws_size;
    const float* x    = (const float*)d_in[0];
    const float* w_ih = (const float*)d_in[1];
    const float* w_hh = (const float*)d_in[2];
    const float* b_ih = (const float*)d_in[3];
    const float* b_hh = (const float*)d_in[4];
    const float* fc_w = (const float*)d_in[5];
    const float* fc_b = (const float*)d_in[6];
    float* out = (float*)d_out;

    // 4096 batches / 16 per wave = 256 waves; 8 waves per 256-thread block -> 32 blocks
    gru_wmma_kernel<<<dim3(32), dim3(256), 0, stream>>>(x, w_ih, w_hh, b_ih, b_hh, fc_w, fc_b, out);
}
